// Wav2Vec2Loss_31353261261423
// MI455X (gfx1250) — matrix operand order repaired and checked by reading, not compiled, not run on hardware
//
#include <hip/hip_runtime.h>
#include <hip/hip_bf16.h>
#include <math.h>

typedef __attribute__((ext_vector_type(16))) _Float16 v16h;
typedef __attribute__((ext_vector_type(8)))  _Float16 v8h;
typedef __attribute__((ext_vector_type(4)))  _Float16 v4h;
typedef __attribute__((ext_vector_type(8)))  float    v8f;
typedef __attribute__((ext_vector_type(4)))  float    v4f;

#define N_ROWS 8192
#define C_DIM  512
#define KSTEPS 16            // 512 / 32 (K per f16 WMMA)
#define COLCHUNKS 8
#define COLS_PER_CHUNK (N_ROWS / COLCHUNKS)      // 1024
#define TILES_PER_CHUNK (COLS_PER_CHUNK / 16)    // 64
#define GSTEPS (TILES_PER_CHUNK * KSTEPS)        // 1024
#define INV_TEMP 10.0f

#define NEG_INF (-__builtin_inff())

// ---------------- Kernel 1: row L2-normalize (fp32 -> f16) ----------------
__global__ __launch_bounds__(256)
void normalize_rows(const float* __restrict__ x, const float* __restrict__ q,
                    _Float16* __restrict__ xn, _Float16* __restrict__ qn) {
  int wave = threadIdx.x >> 5;
  int lane = threadIdx.x & 31;
  int rowg = blockIdx.x * 8 + wave;               // 0 .. 2*N_ROWS-1
  const float* src; _Float16* dst; int r;
  if (rowg < N_ROWS) { src = x; dst = xn; r = rowg; }
  else               { src = q; dst = qn; r = rowg - N_ROWS; }

  const float* p = src + r * C_DIM;
  v4f vals[4];
  float ss = 0.0f;
#pragma unroll
  for (int i = 0; i < 4; ++i) {
    vals[i] = *(const v4f*)(p + i * 128 + lane * 4);
#pragma unroll
    for (int j = 0; j < 4; ++j) ss += vals[i][j] * vals[i][j];
  }
#pragma unroll
  for (int msk = 16; msk >= 1; msk >>= 1) ss += __shfl_xor(ss, msk, 32);

  float inv = 1.0f / fmaxf(sqrtf(ss), 1e-12f);
  _Float16* dp = dst + r * C_DIM;
#pragma unroll
  for (int i = 0; i < 4; ++i) {
    v4h hv;
#pragma unroll
    for (int j = 0; j < 4; ++j) hv[j] = (_Float16)(vals[i][j] * inv);
    *(v4h*)(dp + i * 128 + lane * 4) = hv;
  }
}

// ---------------- Kernel 2: fused WMMA GEMM + streaming softmax -----------
// grid = (64 row-blocks, 8 column-chunks), block = 256 (8 waves).
// A strip (16 rows x 512) register-resident; B stream 4-deep reg pipelined,
// prefetched 2 tiles ahead into WGP$ via global_prefetch.
__global__ __launch_bounds__(256)
void sim_gemm(const _Float16* __restrict__ xn, const _Float16* __restrict__ qn,
              float* __restrict__ partial,   // [row][chunk][4] = m, s, bestv, bestc
              float* __restrict__ diag) {    // [row]
  int wave = threadIdx.x >> 5;
  int lane = threadIdx.x & 31;
  int rowBase = blockIdx.x * 128 + wave * 16;
  int chunkBase = blockIdx.y * COLS_PER_CHUNK;
  int h  = lane >> 4;      // half-wave: 0 or 1
  int rl = lane & 15;

  // A fragments per ISA 16-bit A layout (K halves interleaved at +0 / +16).
  const _Float16* arow = xn + (rowBase + rl) * C_DIM + h * 8;
  v16h A[KSTEPS];
#pragma unroll
  for (int k = 0; k < KSTEPS; ++k) {
    v8h lo = *(const v8h*)(arow + k * 32);
    v8h hi = *(const v8h*)(arow + k * 32 + 16);
    A[k] = __builtin_shufflevector(lo, hi, 0,1,2,3,4,5,6,7,8,9,10,11,12,13,14,15);
  }

  // B fragment source: lane rl = column, K halves interleaved like A (kb+8h+0..7, kb+16+8h+0..7).
  const _Float16* bbase = qn + (chunkBase + rl) * C_DIM + h * 8;
  auto loadB = [&](int g) -> v16h {
    const _Float16* p = bbase + (g >> 4) * (16 * C_DIM) + (g & 15) * 32;
    v8h lo = *(const v8h*)(p);
    v8h hi = *(const v8h*)(p + 16);
    return __builtin_shufflevector(lo, hi, 0,1,2,3,4,5,6,7,8,9,10,11,12,13,14,15);
  };

  // 4-deep rotating register pipeline for the B stream.
  v16h Bf[4];
#pragma unroll
  for (int i = 0; i < 4; ++i) Bf[i] = loadB(i);

  float m[8], s[8], bestv[8];
  int bestc[8];
#pragma unroll
  for (int v = 0; v < 8; ++v) { m[v] = NEG_INF; s[v] = 0.0f; bestv[v] = NEG_INF; bestc[v] = 0; }
  float dval = 0.0f;

  for (int t = 0; t < TILES_PER_CHUNK; ++t) {
    // Prefetch B rows of tile t+2 (~32 K-steps ahead) into WGP$/L2.
    {
      int tn = (t + 2) & (TILES_PER_CHUNK - 1);
      const _Float16* pf = qn + (chunkBase + tn * 16 + rl) * C_DIM;
#pragma unroll
      for (int j = 0; j < 4; ++j)
        __builtin_prefetch(pf + (h * 4 + j) * 64, 0, 3);
    }

    v8f acc = {0.f, 0.f, 0.f, 0.f, 0.f, 0.f, 0.f, 0.f};
#pragma unroll
    for (int k = 0; k < KSTEPS; ++k) {
      int g = t * KSTEPS + k;
      acc = __builtin_amdgcn_wmma_f32_16x16x32_f16(
          /*neg_a=*/false, A[k], /*neg_b=*/false, Bf[g & 3],
          /*c_mod=*/(short)0, acc, /*reuse_a=*/false, /*reuse_b=*/false);
      Bf[g & 3] = loadB((g + 4) & (GSTEPS - 1));   // wrapped tail loads are dead
    }

    // Streaming softmax / argmax update. acc[v]: row=rowBase+v+8h, col=colBase+rl
    int col = chunkBase + t * 16 + rl;
#pragma unroll
    for (int v = 0; v < 8; ++v) {
      float val = acc[v] * INV_TEMP;
      float nm = fmaxf(m[v], val);
      s[v] = s[v] * __expf(m[v] - nm) + __expf(val - nm);
      m[v] = nm;
      bool better = val > bestv[v];
      bestc[v] = better ? col : bestc[v];
      bestv[v] = better ? val : bestv[v];
      int row = rowBase + v + 8 * h;
      dval = (col == row) ? val : dval;     // diagonal capture (branchless)
    }
  }

  // Merge the 16 column-lanes within each half-wave (masks 1..8 stay in half).
#pragma unroll
  for (int v = 0; v < 8; ++v) {
#pragma unroll
    for (int msk = 8; msk >= 1; msk >>= 1) {
      float om = __shfl_xor(m[v], msk, 32);
      float os = __shfl_xor(s[v], msk, 32);
      float nm = fmaxf(m[v], om);
      s[v] = s[v] * __expf(m[v] - nm) + os * __expf(om - nm);
      m[v] = nm;
      float obv = __shfl_xor(bestv[v], msk, 32);
      int   obc = __shfl_xor(bestc[v], msk, 32);
      bool take = (obv > bestv[v]) || ((obv == bestv[v]) && (obc < bestc[v]));
      bestv[v] = take ? obv : bestv[v];
      bestc[v] = take ? obc : bestc[v];
    }
  }

  // Lane (rl==v) in each half writes the partial state for row rowBase+v+8h.
#pragma unroll
  for (int v = 0; v < 8; ++v) {
    if (rl == v) {
      int row = rowBase + v + 8 * h;
      float* pp = partial + (row * COLCHUNKS + blockIdx.y) * 4;
      pp[0] = m[v];
      pp[1] = s[v];
      pp[2] = bestv[v];
      pp[3] = __int_as_float(bestc[v]);
    }
  }
  // Diagonal: lane with rl == v+8h (i.e. lanes 0-7 and 24-31) owns one row.
  {
    int v = rl - 8 * h;
    int myrow = rowBase + rl;
    if (v >= 0 && v < 8 && myrow >= chunkBase && myrow < chunkBase + COLS_PER_CHUNK)
      diag[myrow] = dval;
  }
}

// ---------------- Kernel 3: per-row merge of chunk partials ----------------
__global__ __launch_bounds__(256)
void merge_rows(const float* __restrict__ partial, const float* __restrict__ diag,
                float* __restrict__ partial2) {
  __shared__ float sl[256];
  __shared__ float sa[256];
  int tid = threadIdx.x;
  int r = blockIdx.x * 256 + tid;

  float m = NEG_INF, s = 0.0f, bv = NEG_INF;
  int bc = 0;
#pragma unroll
  for (int c = 0; c < COLCHUNKS; ++c) {
    const float* pp = partial + (r * COLCHUNKS + c) * 4;
    float om = pp[0], os = pp[1], obv = pp[2];
    int obc = __float_as_int(pp[3]);
    float nm = fmaxf(m, om);
    s = s * __expf(m - nm) + os * __expf(om - nm);
    m = nm;
    bool take = (obv > bv) || ((obv == bv) && (obc < bc));
    bv = take ? obv : bv;
    bc = take ? obc : bc;
  }
  float loss = m + logf(s) - diag[r];       // logsumexp - sim_ii
  float acc  = (bc == r) ? 1.0f : 0.0f;

  sl[tid] = loss; sa[tid] = acc;
  __syncthreads();
  for (int off = 128; off > 0; off >>= 1) {
    if (tid < off) { sl[tid] += sl[tid + off]; sa[tid] += sa[tid + off]; }
    __syncthreads();
  }
  if (tid == 0) {
    partial2[blockIdx.x * 2 + 0] = sl[0];
    partial2[blockIdx.x * 2 + 1] = sa[0];
  }
}

// ---------------- Kernel 4: final reduce (32 block partials) ---------------
__global__ void finalize(const float* __restrict__ partial2, float* __restrict__ out) {
  int lane = threadIdx.x & 31;
  float l = partial2[lane * 2 + 0];
  float a = partial2[lane * 2 + 1];
#pragma unroll
  for (int msk = 16; msk >= 1; msk >>= 1) {
    l += __shfl_xor(l, msk, 32);
    a += __shfl_xor(a, msk, 32);
  }
  if (lane == 0) {
    out[0] = l * (1.0f / (float)N_ROWS);
    out[1] = a * (1.0f / (float)N_ROWS);
  }
}

// ---------------- Host-side launch -----------------------------------------
extern "C" void kernel_launch(void* const* d_in, const int* in_sizes, int n_in,
                              void* d_out, int out_size, void* d_ws, size_t ws_size,
                              hipStream_t stream) {
  const float* x = (const float*)d_in[0];
  const float* q = (const float*)d_in[1];
  float* out = (float*)d_out;

  char* ws = (char*)d_ws;
  const size_t XN_BYTES = (size_t)N_ROWS * C_DIM * sizeof(_Float16);  // 8 MB
  _Float16* xn   = (_Float16*)(ws);
  _Float16* qn   = (_Float16*)(ws + XN_BYTES);
  float*    diag = (float*)   (ws + 2 * XN_BYTES);                       // 32 KB
  float*    part = (float*)   (ws + 2 * XN_BYTES + (size_t)N_ROWS * 4);  // 1 MB
  float*    part2= (float*)   (ws + 2 * XN_BYTES + (size_t)N_ROWS * 4
                                  + (size_t)N_ROWS * COLCHUNKS * 4 * 4); // 256 B

  normalize_rows<<<(2 * N_ROWS) / 8, 256, 0, stream>>>(x, q, xn, qn);
  sim_gemm<<<dim3(N_ROWS / 128, COLCHUNKS), 256, 0, stream>>>(xn, qn, part, diag);
  merge_rows<<<N_ROWS / 256, 256, 0, stream>>>(part, diag, part2);
  finalize<<<1, 32, 0, stream>>>(part2, out);
}
